// GAT_17703855194739
// MI455X (gfx1250) — matrix operand order repaired and checked by reading, not compiled, not run on hardware
//
#include <hip/hip_runtime.h>
#include <hip/hip_bf16.h>
#include <math.h>

#define N_NODES   100000
#define N_EDGES   1600000
#define IN_CH     128
#define OUT_CH    256
#define HEADS     8
#define HEAD_DIM  32
#define E_TOT     (N_EDGES + N_NODES)

typedef __attribute__((ext_vector_type(16))) _Float16 v16h;
typedef __attribute__((ext_vector_type(8)))  _Float16 v8h;
typedef __attribute__((ext_vector_type(8)))  float    v8f;

// ---------------- helpers ----------------

// monotone float<->uint encoding for atomic max on floats
__device__ __forceinline__ unsigned encF(float f) {
    unsigned u = __float_as_uint(f);
    return (u & 0x80000000u) ? ~u : (u | 0x80000000u);
}
__device__ __forceinline__ float decF(unsigned u) {
    u = (u & 0x80000000u) ? (u & 0x7FFFFFFFu) : ~u;
    return __uint_as_float(u);
}

// ---------------- init ----------------

__global__ void init_kernel(float* __restrict__ accum, unsigned* __restrict__ mEnc,
                            float* __restrict__ denom, float* __restrict__ gsum) {
    size_t i = (size_t)blockIdx.x * blockDim.x + threadIdx.x;
    if (i < (size_t)N_NODES * OUT_CH) accum[i] = 0.0f;
    if (i < (size_t)N_NODES * HEADS) { mEnc[i] = 0u; denom[i] = 0.0f; }
    if (i < OUT_CH) gsum[i] = 0.0f;
}

// ---------------- f32 -> f16 conversion passes ----------------

__global__ void cvt_f16(const float* __restrict__ in, _Float16* __restrict__ out, int n) {
    int i = blockIdx.x * blockDim.x + threadIdx.x;
    if (i < n) out[i] = (_Float16)in[i];
}

// Wcat16[j*128 + k] = W[h=j>>5][k][d=j&31] as f16  (column j contiguous over k)
__global__ void build_wcat(const float* __restrict__ W, _Float16* __restrict__ b) {
    int i = blockIdx.x * blockDim.x + threadIdx.x;
    if (i >= OUT_CH * IN_CH) return;
    const int j = i / IN_CH, k = i % IN_CH;
    b[i] = (_Float16)W[(((size_t)(j >> 5) * IN_CH + k) * HEAD_DIM) + (j & 31)];
}

// ---------------- WMMA GEMM: out[M,256] = A[M,K](f16) @ B(f16, col-major) + bias --
// B layout: B[j*K + k] (column j contiguous over k).
// Block = 10 waves (320 thr): all waves share one 64-column B panel staged in LDS
// via global_load_async_to_lds_b128; each wave owns one 16x64 output tile.
template<int K>
__global__ __launch_bounds__(320)
void gemm_h(const _Float16* __restrict__ A, const _Float16* __restrict__ B,
            const float* __restrict__ bias, float* __restrict__ out) {
    __shared__ _Float16 bpanel[64 * K];               // 16KB (K=128) / 32KB (K=256)

    const int tid  = threadIdx.x;
    const int lane = tid & 31;
    const int wid  = tid >> 5;                        // 0..9
    const int lc   = lane & 15;
    const int half = lane >> 4;
    const int n0   = blockIdx.y * 64;                 // column group
    const int mT   = blockIdx.x * 10 + wid;           // grid exact: no bounds check
    const int row  = mT * 16 + lc;

    // ---- stage B panel into LDS with async copies (16B chunks, 1 per lane) ----
    constexpr int KG = K / 8;                         // 16B chunks per column
    for (int c = tid; c < 64 * KG; c += 320) {
        const int col = c / KG, kg = c % KG;
        unsigned ldsoff = (unsigned)(unsigned long long)(&bpanel[col * K + kg * 8]);
        unsigned long long ga =
            (unsigned long long)(B + (size_t)(n0 + col) * K + kg * 8);
        asm volatile("global_load_async_to_lds_b128 %0, %1, off"
                     :: "v"(ldsoff), "v"(ga) : "memory");
    }
    asm volatile("s_wait_asynccnt 0" ::: "memory");
    __syncthreads();

    // ---- compute 16x64 tile: 4 WMMA accumulators ----
    v8f acc[4];
#pragma unroll
    for (int t = 0; t < 4; ++t)
#pragma unroll
        for (int r = 0; r < 8; ++r) acc[t][r] = 0.0f;

    const v8h* pa = (const v8h*)(A + (size_t)row * K);
    for (int kk = 0; kk < K; kk += 32) {
        // A fragment (16x32 f16): lane = row, two contiguous 8-half groups
        const v8h g0 = pa[(kk >> 3) + half];          // K = kk + half*8 + [0..7]
        const v8h g1 = pa[(kk >> 3) + 2 + half];      // K = kk+16 + half*8 + [0..7]
        const v16h a = __builtin_shufflevector(g0, g1,
            0, 1, 2, 3, 4, 5, 6, 7, 8, 9, 10, 11, 12, 13, 14, 15);
#pragma unroll
        for (int t = 0; t < 4; ++t) {
            // B fragment (32x16 f16): lane = col, K = kk + half*8 + [0..7], kk+16 + half*8 + [0..7]
            const v8h* pb = (const v8h*)&bpanel[(t * 16 + lc) * K + kk];
            const v16h b = __builtin_shufflevector(pb[half], pb[2 + half],
                0, 1, 2, 3, 4, 5, 6, 7, 8, 9, 10, 11, 12, 13, 14, 15);
            acc[t] = __builtin_amdgcn_wmma_f32_16x16x32_f16(
                false, a, false, b, (short)0, acc[t], false, false);
        }
    }
#pragma unroll
    for (int t = 0; t < 4; ++t) {
        const int col = n0 + t * 16 + lc;
        const float bv = bias ? bias[col] : 0.0f;
#pragma unroll
        for (int r = 0; r < 8; ++r) {
            const int m = mT * 16 + r + 8 * half;
            out[(size_t)m * OUT_CH + col] = acc[t][r] + bv;
        }
    }
}

// ---------------- attention dot products ----------------
__global__ void attn_dots(const float* __restrict__ xh,
                          const float* __restrict__ att_src,
                          const float* __restrict__ att_dst,
                          float* __restrict__ asrc, float* __restrict__ adst) {
    int i = blockIdx.x * blockDim.x + threadIdx.x; // node*8 + head
    if (i >= N_NODES * HEADS) return;
    const int h = i & 7;
    const float* row = xh + (size_t)(i >> 3) * OUT_CH + h * HEAD_DIM;
    float s = 0.0f, d2 = 0.0f;
#pragma unroll
    for (int d = 0; d < HEAD_DIM; ++d) {
        const float v = row[d];
        s  += v * att_src[h * HEAD_DIM + d];
        d2 += v * att_dst[h * HEAD_DIM + d];
    }
    asrc[i] = s;
    adst[i] = d2;
}

// ---------------- edge phase ----------------
__device__ __forceinline__ void edge_ends(const int* ei, int e, int& s, int& d) {
    if (e < N_EDGES) { s = ei[e]; d = ei[N_EDGES + e]; }
    else             { s = d = e - N_EDGES; }   // self loop
}

__global__ void edge_max(const int* __restrict__ ei, const float* __restrict__ asrc,
                         const float* __restrict__ adst, unsigned* __restrict__ mEnc) {
    int idx = blockIdx.x * blockDim.x + threadIdx.x;
    if (idx >= E_TOT * HEADS) return;
    const int e = idx >> 3, h = idx & 7;
    int s, d; edge_ends(ei, e, s, d);
    float v = asrc[s * HEADS + h] + adst[d * HEADS + h];
    v = (v >= 0.0f) ? v : 0.2f * v;
    atomicMax(&mEnc[d * HEADS + h], encF(v));
}

__global__ void edge_sum(const int* __restrict__ ei, const float* __restrict__ asrc,
                         const float* __restrict__ adst, const unsigned* __restrict__ mEnc,
                         float* __restrict__ denom) {
    int idx = blockIdx.x * blockDim.x + threadIdx.x;
    if (idx >= E_TOT * HEADS) return;
    const int e = idx >> 3, h = idx & 7;
    int s, d; edge_ends(ei, e, s, d);
    float v = asrc[s * HEADS + h] + adst[d * HEADS + h];
    v = (v >= 0.0f) ? v : 0.2f * v;
    atomicAdd(&denom[d * HEADS + h], __expf(v - decF(mEnc[d * HEADS + h])));
}

// one 256-thread block per edge: coalesced row gather + coalesced atomic scatter
__global__ __launch_bounds__(256)
void edge_agg(const int* __restrict__ ei, const float* __restrict__ asrc,
              const float* __restrict__ adst, const unsigned* __restrict__ mEnc,
              const float* __restrict__ denom, const float* __restrict__ xh,
              float* __restrict__ accum) {
    __shared__ float alpha_s[HEADS];
    const int e = blockIdx.x;
    int s, d; edge_ends(ei, e, s, d);
    const int t = threadIdx.x;
    if (t < HEADS) {
        float v = asrc[s * HEADS + t] + adst[d * HEADS + t];
        v = (v >= 0.0f) ? v : 0.2f * v;
        const float ex = __expf(v - decF(mEnc[d * HEADS + t]));
        alpha_s[t] = ex / denom[d * HEADS + t];
    }
    __syncthreads();
    const float a = alpha_s[t >> 5];
    atomicAdd(&accum[(size_t)d * OUT_CH + t], xh[(size_t)s * OUT_CH + t] * a);
}

__global__ void bias_add(float* __restrict__ accum, const float* __restrict__ conv_bias) {
    size_t i = (size_t)blockIdx.x * blockDim.x + threadIdx.x;
    if (i < (size_t)N_NODES * OUT_CH) accum[i] += conv_bias[i & (OUT_CH - 1)];
}

// ---------------- rowwise: sa = softmax(leaky(z,.01)); z <- leaky(xl*sa,.2) ------
__global__ __launch_bounds__(256)
void row_sa(float* __restrict__ z, const float* __restrict__ xl) {
    __shared__ float red[OUT_CH];
    const size_t n = blockIdx.x;
    const int t = threadIdx.x;
    float v = z[n * OUT_CH + t];
    v = (v >= 0.0f) ? v : 0.01f * v;
    red[t] = v; __syncthreads();
    for (int s = 128; s > 0; s >>= 1) {
        if (t < s) red[t] = fmaxf(red[t], red[t + s]);
        __syncthreads();
    }
    const float m = red[0]; __syncthreads();
    const float ex = __expf(v - m);
    red[t] = ex; __syncthreads();
    for (int s = 128; s > 0; s >>= 1) {
        if (t < s) red[t] += red[t + s];
        __syncthreads();
    }
    const float sa = ex / red[0];
    float tv = xl[n * OUT_CH + t] * sa;
    tv = (tv >= 0.0f) ? tv : 0.2f * tv;
    z[n * OUT_CH + t] = tv;
}

// ---------------- rowwise LayerNorm + L2 normalize + global mean accumulation ----
__global__ __launch_bounds__(256)
void row_ln(float* __restrict__ y, const float* __restrict__ lng,
            const float* __restrict__ lnb, float* __restrict__ gsum) {
    __shared__ float red[OUT_CH];
    const int t = threadIdx.x;
    float acc = 0.0f;
    for (size_t n = blockIdx.x; n < N_NODES; n += gridDim.x) {
        const float v = y[n * OUT_CH + t];
        red[t] = v; __syncthreads();
        for (int s = 128; s > 0; s >>= 1) { if (t < s) red[t] += red[t + s]; __syncthreads(); }
        const float mu = red[0] * (1.0f / OUT_CH); __syncthreads();
        const float c = v - mu;
        red[t] = c * c; __syncthreads();
        for (int s = 128; s > 0; s >>= 1) { if (t < s) red[t] += red[t + s]; __syncthreads(); }
        const float var = red[0] * (1.0f / OUT_CH); __syncthreads();
        float o = c * rsqrtf(var + 1e-5f) * lng[t] + lnb[t];
        red[t] = o * o; __syncthreads();
        for (int s = 128; s > 0; s >>= 1) { if (t < s) red[t] += red[t + s]; __syncthreads(); }
        const float nrm = fmaxf(sqrtf(red[0]), 1e-12f); __syncthreads();
        o /= nrm;
        y[n * OUT_CH + t] = o;
        acc += o;
    }
    atomicAdd(&gsum[t], acc);
}

// ---------------- global gate: ga = softmax(relu(g @ gfc_w^T + gfc_b)) -----------
__global__ __launch_bounds__(256)
void global_gate(const float* __restrict__ gsum, const float* __restrict__ gw,
                 const float* __restrict__ gb, float* __restrict__ ga) {
    __shared__ float g[OUT_CH];
    __shared__ float red[OUT_CH];
    const int t = threadIdx.x;
    g[t] = gsum[t] * (1.0f / N_NODES);
    __syncthreads();
    float acc = gb[t];
    for (int k = 0; k < OUT_CH; ++k) acc += gw[(size_t)t * OUT_CH + k] * g[k];
    acc = fmaxf(acc, 0.0f);
    red[t] = acc; __syncthreads();
    for (int s = 128; s > 0; s >>= 1) { if (t < s) red[t] = fmaxf(red[t], red[t + s]); __syncthreads(); }
    const float m = red[0]; __syncthreads();
    const float ex = __expf(acc - m);
    red[t] = ex; __syncthreads();
    for (int s = 128; s > 0; s >>= 1) { if (t < s) red[t] += red[t + s]; __syncthreads(); }
    ga[t] = ex / red[0];
}

__global__ void scale_out(float* __restrict__ y, const float* __restrict__ ga) {
    size_t i = (size_t)blockIdx.x * blockDim.x + threadIdx.x;
    if (i < (size_t)N_NODES * OUT_CH) y[i] *= ga[i & (OUT_CH - 1)];
}

// ---------------- launch ----------------
extern "C" void kernel_launch(void* const* d_in, const int* in_sizes, int n_in,
                              void* d_out, int out_size, void* d_ws, size_t ws_size,
                              hipStream_t stream) {
    const float* x         = (const float*)d_in[0];
    const int*   ei        = (const int*)  d_in[1];
    // d_in[2] = edge_attr (unused: edge_dim=None in reference)
    const float* W         = (const float*)d_in[3];
    const float* att_src   = (const float*)d_in[4];
    const float* att_dst   = (const float*)d_in[5];
    const float* conv_bias = (const float*)d_in[6];
    const float* fc_w      = (const float*)d_in[7];
    const float* fc_b      = (const float*)d_in[8];
    const float* ln_g      = (const float*)d_in[9];
    const float* ln_b      = (const float*)d_in[10];
    const float* gfc_w     = (const float*)d_in[11];
    const float* gfc_b     = (const float*)d_in[12];

    float* out = (float*)d_out;                    // xl accumulator, later final output

    // workspace layout
    float* ws = (float*)d_ws;
    float*    xh    = ws;                                          // [N,256] f32; z1 alias
    float*    asrc  = xh   + (size_t)N_NODES * OUT_CH;             // [N,8]
    float*    adst  = asrc + (size_t)N_NODES * HEADS;              // [N,8]
    unsigned* mEnc  = (unsigned*)(adst + (size_t)N_NODES * HEADS); // [N,8]
    float*    denom = (float*)mEnc + (size_t)N_NODES * HEADS;      // [N,8]
    float*    gsum  = denom + (size_t)N_NODES * HEADS;             // [256]
    float*    ga    = gsum + OUT_CH;                               // [256]
    _Float16* a16   = (_Float16*)(ga + OUT_CH);                    // [N,256] f16 (also x16)
    _Float16* wcat16= a16 + (size_t)N_NODES * OUT_CH;              // [256*128] f16
    _Float16* fcw16 = wcat16 + OUT_CH * IN_CH;                     // [256*256] f16
    float*    z1    = xh;   // xh dead after edge_agg; reuse for z1/t
    _Float16* x16   = a16;  // a16 region hosts x16 first (dead before a16 written)

    const dim3 gemmGrid(N_NODES / 16 / 10, OUT_CH / 64);           // 625 x 4, 320 thr
    const int elemBlocks = (N_NODES * OUT_CH + 255) / 256;         // 100000
    const int nhBlocks   = (N_NODES * HEADS + 255) / 256;          // 3125
    const int ehBlocks   = (E_TOT * HEADS + 255) / 256;            // 53125
    const int xBlocks    = (N_NODES * IN_CH + 255) / 256;          // 50000

    init_kernel<<<elemBlocks, 256, 0, stream>>>(out, mEnc, denom, gsum);
    cvt_f16<<<xBlocks, 256, 0, stream>>>(x, x16, N_NODES * IN_CH);
    build_wcat<<<(OUT_CH * IN_CH + 255) / 256, 256, 0, stream>>>(W, wcat16);
    cvt_f16<<<(OUT_CH * OUT_CH + 255) / 256, 256, 0, stream>>>(fc_w, fcw16, OUT_CH * OUT_CH);

    gemm_h<IN_CH><<<gemmGrid, 320, 0, stream>>>(x16, wcat16, nullptr, xh);

    attn_dots<<<nhBlocks, 256, 0, stream>>>(xh, att_src, att_dst, asrc, adst);
    edge_max<<<ehBlocks, 256, 0, stream>>>(ei, asrc, adst, mEnc);
    edge_sum<<<ehBlocks, 256, 0, stream>>>(ei, asrc, adst, mEnc, denom);
    edge_agg<<<E_TOT, 256, 0, stream>>>(ei, asrc, adst, mEnc, denom, xh, out);
    bias_add<<<elemBlocks, 256, 0, stream>>>(out, conv_bias);

    cvt_f16<<<elemBlocks, 256, 0, stream>>>(out, a16, N_NODES * OUT_CH);
    gemm_h<OUT_CH><<<gemmGrid, 320, 0, stream>>>(a16, fcw16, fc_b, z1);   // z1 = xl@fcW^T+b
    row_sa<<<N_NODES, 256, 0, stream>>>(z1, out);                         // z1 <- leaky(xl*sa)
    cvt_f16<<<elemBlocks, 256, 0, stream>>>(z1, a16, N_NODES * OUT_CH);
    gemm_h<OUT_CH><<<gemmGrid, 320, 0, stream>>>(a16, fcw16, fc_b, out);  // out = t@fcW^T+b

    row_ln<<<2048, 256, 0, stream>>>(out, ln_g, ln_b, gsum);
    global_gate<<<1, 256, 0, stream>>>(gsum, gfc_w, gfc_b, ga);
    scale_out<<<elemBlocks, 256, 0, stream>>>(out, ga);
}
